// ManualMultiheadAttention_33328946217775
// MI455X (gfx1250) — matrix operand order repaired and checked by reading, not compiled, not run on hardware
//
#include <hip/hip_runtime.h>
#include <hip/hip_bf16.h>

// ---------------------------------------------------------------------------
// Manual multi-head attention, CDNA5 (gfx1250), bf16 WMMA pipeline.
// B=4, S=2048, E=1024, H=16, D=64.
// ---------------------------------------------------------------------------

typedef __attribute__((ext_vector_type(16))) __bf16 v16bf;
typedef __attribute__((ext_vector_type(8)))  __bf16 v8bf;
typedef __attribute__((ext_vector_type(8)))  float  v8f;

__device__ __forceinline__ __bf16 f2bf(float f) {
  unsigned u = __builtin_bit_cast(unsigned, f);
  u += 0x7fffu + ((u >> 16) & 1u);            // round-to-nearest-even
  unsigned short h = (unsigned short)(u >> 16);
  return __builtin_bit_cast(__bf16, h);
}

__device__ __forceinline__ v16bf cat16(v8bf lo, v8bf hi) {
  return __builtin_shufflevector(lo, hi, 0,1,2,3,4,5,6,7,8,9,10,11,12,13,14,15);
}

__device__ __forceinline__ v8f wmma_bf16(v16bf a, v16bf b, v8f c) {
  // D = A(16x32) * B(32x16) + C(16x16), f32 accumulate
  return __builtin_amdgcn_wmma_f32_16x16x32_bf16(false, a, false, b, (short)0, c,
                                                 false, false);
}

__device__ __forceinline__ float redmax16(float t) {
#pragma unroll
  for (int m = 1; m <= 8; m <<= 1) t = fmaxf(t, __shfl_xor(t, m, 32));
  return t;
}
__device__ __forceinline__ float redsum16(float t) {
#pragma unroll
  for (int m = 1; m <= 8; m <<= 1) t += __shfl_xor(t, m, 32);
  return t;
}

// ---------------------------------------------------------------------------
// fp32 -> bf16 conversion, 4 elements / thread
// ---------------------------------------------------------------------------
__global__ __launch_bounds__(256) void cvt_kernel(const float* __restrict__ in,
                                                  __bf16* __restrict__ out,
                                                  int n4) {
  int i = blockIdx.x * blockDim.x + threadIdx.x;
  if (i >= n4) return;
  float4 f = ((const float4*)in)[i];
  unsigned lo = ((unsigned)__builtin_bit_cast(unsigned short, f2bf(f.y)) << 16) |
                (unsigned)__builtin_bit_cast(unsigned short, f2bf(f.x));
  unsigned hi = ((unsigned)__builtin_bit_cast(unsigned short, f2bf(f.w)) << 16) |
                (unsigned)__builtin_bit_cast(unsigned short, f2bf(f.z));
  ((uint2*)out)[i] = make_uint2(lo, hi);
}

// ---------------------------------------------------------------------------
// Shared GEMM mainloop structure (double-buffered LDS, 1 barrier / K-step):
// block tile 256x128, K-step 32, 8 waves as 4x2 grid of 64x64 wave tiles.
// Per K-step per wave: 4 A-frags + 4 B-frags (16 ds_load_b128), 16 WMMAs.
// A tile staged 256x32 (1 row / thread), B tile 128x32 (half row / thread).
// ---------------------------------------------------------------------------
#define GEMM_LOAD_TILE(k0)                                                    \
  {                                                                           \
    const __bf16* ga = A + (size_t)(m0 + tid) * 1024 + (k0);                  \
    ra0 = *(const v8bf*)ga;        ra1 = *(const v8bf*)(ga + 8);              \
    ra2 = *(const v8bf*)(ga + 16); ra3 = *(const v8bf*)(ga + 24);             \
    const __bf16* gb =                                                        \
        W + (size_t)(n0 + (tid >> 1)) * 1024 + (k0) + (tid & 1) * 16;         \
    rb0 = *(const v8bf*)gb;        rb1 = *(const v8bf*)(gb + 8);              \
  }

#define GEMM_MAINLOOP()                                                       \
  v8bf ra0, ra1, ra2, ra3, rb0, rb1;                                          \
  GEMM_LOAD_TILE(0)                                                           \
  for (int it = 0; it < 32; ++it) {                                           \
    __bf16* sA = &lA[it & 1][0];                                              \
    __bf16* sB = &lB[it & 1][0];                                              \
    *(v8bf*)&sA[tid * 40 + 0]  = ra0;                                         \
    *(v8bf*)&sA[tid * 40 + 8]  = ra1;                                         \
    *(v8bf*)&sA[tid * 40 + 16] = ra2;                                         \
    *(v8bf*)&sA[tid * 40 + 24] = ra3;                                         \
    *(v8bf*)&sB[(tid >> 1) * 40 + (tid & 1) * 16 + 0] = rb0;                  \
    *(v8bf*)&sB[(tid >> 1) * 40 + (tid & 1) * 16 + 8] = rb1;                  \
    __syncthreads();                                                          \
    if (it + 1 < 32) GEMM_LOAD_TILE((it + 1) * 32)                            \
    v16bf af[4];                                                              \
    _Pragma("unroll")                                                         \
    for (int rt = 0; rt < 4; ++rt) {                                          \
      int ar = wr + rt * 16 + lm;                                             \
      af[rt] = cat16(*(const v8bf*)&sA[ar * 40 + hi * 8],                     \
                     *(const v8bf*)&sA[ar * 40 + hi * 8 + 16]);               \
    }                                                                         \
    _Pragma("unroll")                                                         \
    for (int nt = 0; nt < 4; ++nt) {                                          \
      int br = wc + nt * 16 + lm;                                             \
      v16bf bfm = cat16(*(const v8bf*)&sB[br * 40 + hi * 8],                  \
                        *(const v8bf*)&sB[br * 40 + hi * 8 + 16]);            \
      _Pragma("unroll")                                                       \
      for (int rt = 0; rt < 4; ++rt)                                          \
        acc[rt][nt] = wmma_bf16(af[rt], bfm, acc[rt][nt]);                    \
    }                                                                         \
  }

// ---------------------------------------------------------------------------
// QKV projection: C[8192, 3072] = X[8192,1024] @ W[3072,1024]^T + bias
// Epilogue scatters bf16 into Q,K: [B,H,S,D] and V transposed: [B,H,D,S].
// ---------------------------------------------------------------------------
__global__ __launch_bounds__(256) void gemm_qkv_kernel(
    const __bf16* __restrict__ A, const __bf16* __restrict__ W,
    const float* __restrict__ bias, __bf16* __restrict__ q,
    __bf16* __restrict__ kk, __bf16* __restrict__ vT) {
  __shared__ __bf16 lA[2][256 * 40];
  __shared__ __bf16 lB[2][128 * 40];

  const int m0 = blockIdx.x * 256;
  const int n0 = blockIdx.y * 128;
  const int tid = threadIdx.x;
  const int wave = tid >> 5, lane = tid & 31;
  const int lm = lane & 15, hi = lane >> 4;
  const int wr = (wave >> 1) * 64, wc = (wave & 1) * 64;

  v8f acc[4][4] = {};
  GEMM_MAINLOOP()

#pragma unroll
  for (int rt = 0; rt < 4; ++rt) {
#pragma unroll
    for (int nt = 0; nt < 4; ++nt) {
      int n = n0 + wc + nt * 16 + lm;
      float bv = bias[n];
      int which = n >> 10;            // 0:q 1:k 2:v
      int e = n & 1023;
      int h = e >> 6, d = e & 63;
#pragma unroll
      for (int r = 0; r < 8; ++r) {
        int m = m0 + wr + rt * 16 + hi * 8 + r;   // C layout: VGPR r, half-wave
        int b = m >> 11, s = m & 2047;
        __bf16 val = f2bf(acc[rt][nt][r] + bv);
        size_t bh = (size_t)(b * 16 + h);
        if (which == 0)      q[(bh * 2048 + s) * 64 + d] = val;
        else if (which == 1) kk[(bh * 2048 + s) * 64 + d] = val;
        else                 vT[(bh * 64 + d) * 2048 + s] = val;
      }
    }
  }
}

// ---------------------------------------------------------------------------
// Output projection: out[8192,1024] = AO[8192,1024] @ Wo[1024,1024]^T + bias
// ---------------------------------------------------------------------------
__global__ __launch_bounds__(256) void gemm_out_kernel(
    const __bf16* __restrict__ A, const __bf16* __restrict__ W,
    const float* __restrict__ bias, float* __restrict__ out) {
  __shared__ __bf16 lA[2][256 * 40];
  __shared__ __bf16 lB[2][128 * 40];

  const int m0 = blockIdx.x * 256;
  const int n0 = blockIdx.y * 128;
  const int tid = threadIdx.x;
  const int wave = tid >> 5, lane = tid & 31;
  const int lm = lane & 15, hi = lane >> 4;
  const int wr = (wave >> 1) * 64, wc = (wave & 1) * 64;

  v8f acc[4][4] = {};
  GEMM_MAINLOOP()

#pragma unroll
  for (int rt = 0; rt < 4; ++rt) {
#pragma unroll
    for (int nt = 0; nt < 4; ++nt) {
      int n = n0 + wc + nt * 16 + lm;
      float bv = bias[n];
#pragma unroll
      for (int r = 0; r < 8; ++r) {
        int m = m0 + wr + rt * 16 + hi * 8 + r;
        out[(size_t)m * 1024 + n] = acc[rt][nt][r] + bv;
      }
    }
  }
}

// ---------------------------------------------------------------------------
// Flash attention per (b,h): block = 4 waves, each wave owns 16 query rows,
// key tile = 64 (8 QK^T WMMAs + softmax + 8 PV WMMAs per iteration).
// Q,K: [B,H,S,64] bf16 ; V transposed: [B,H,64,S] bf16 ; out: [B,S,E] bf16.
// ---------------------------------------------------------------------------
__global__ __launch_bounds__(128) void attn_kernel(
    const __bf16* __restrict__ Q, const __bf16* __restrict__ K,
    const __bf16* __restrict__ Vt, __bf16* __restrict__ Ao) {
  __shared__ __bf16 lP[4][16 * 72];   // per-wave 16x64 P tile (stride 72)

  const int bh = blockIdx.y;
  const int wave = threadIdx.x >> 5, lane = threadIdx.x & 31;
  const int lm = lane & 15, hi = lane >> 4;
  const int q0 = blockIdx.x * 64 + wave * 16;

  const __bf16* Qp = Q + (size_t)bh * (2048 * 64);
  const __bf16* Kp = K + (size_t)bh * (2048 * 64);
  const __bf16* Vp = Vt + (size_t)bh * (64 * 2048);

  const int ka = hi * 8;     // A-fragment K sub-offset
  const int kb2 = hi * 8;    // B-fragment K sub-offset

  // Q fragments: 16 rows x 64 d  -> two 16x32 A-frags, kept in registers
  v16bf aq[2];
#pragma unroll
  for (int c = 0; c < 2; ++c) {
    const __bf16* p = Qp + (size_t)(q0 + lm) * 64 + c * 32 + ka;
    aq[c] = cat16(*(const v8bf*)p, *(const v8bf*)(p + 16));
  }

  v8f o[4] = {};               // O accumulator: 16 rows x 64 d
  float mrow[8], lrow[8];
#pragma unroll
  for (int r = 0; r < 8; ++r) { mrow[r] = -1e30f; lrow[r] = 0.f; }

  __bf16* pP = &lP[wave][0];

  for (int t0 = 0; t0 < 2048; t0 += 64) {
    // scores S = Q K^T  (four 16x16 tiles over 64 keys)
    v8f sv[4] = {};
#pragma unroll
    for (int nt = 0; nt < 4; ++nt) {
      const __bf16* kp = Kp + (size_t)(t0 + nt * 16 + lm) * 64 + kb2;
#pragma unroll
      for (int c = 0; c < 2; ++c) {
        v16bf bk = cat16(*(const v8bf*)(kp + c * 32),
                         *(const v8bf*)(kp + c * 32 + 16));
        sv[nt] = wmma_bf16(aq[c], bk, sv[nt]);
      }
    }

    if (t0 + 64 < 2048) {   // warm next K/V tile toward L0 (L2-resident data)
      __builtin_prefetch(Kp + (size_t)(t0 + 64 + lm) * 64, 0, 0);
      __builtin_prefetch(Vp + (size_t)lm * 2048 + t0 + 64, 0, 0);
    }

    // online softmax across the 64-key tile, write P (bf16) to per-wave LDS
#pragma unroll
    for (int r = 0; r < 8; ++r) {
      float s0 = sv[0][r] * 0.125f;           // 1/sqrt(64)
      float s1 = sv[1][r] * 0.125f;
      float s2 = sv[2][r] * 0.125f;
      float s3 = sv[3][r] * 0.125f;
      float mt = redmax16(fmaxf(fmaxf(s0, s1), fmaxf(s2, s3)));
      float mnew = fmaxf(mrow[r], mt);
      float p0 = __expf(s0 - mnew);
      float p1 = __expf(s1 - mnew);
      float p2 = __expf(s2 - mnew);
      float p3 = __expf(s3 - mnew);
      float rs = redsum16((p0 + p1) + (p2 + p3));
      float ef = __expf(mrow[r] - mnew);
      lrow[r] = lrow[r] * ef + rs;
      mrow[r] = mnew;
#pragma unroll
      for (int dt = 0; dt < 4; ++dt) o[dt][r] *= ef;
      int prow = hi * 8 + r;
      pP[prow * 72 + lm]      = f2bf(p0);     // cols  0..15
      pP[prow * 72 + 16 + lm] = f2bf(p1);     // cols 16..31
      pP[prow * 72 + 32 + lm] = f2bf(p2);     // cols 32..47
      pP[prow * 72 + 48 + lm] = f2bf(p3);     // cols 48..63
    }
    asm volatile("s_wait_dscnt 0" ::: "memory");

    // reload P as two 16x32 A-fragments (K = 64 keys)
    v16bf ap[2];
#pragma unroll
    for (int c = 0; c < 2; ++c)
      ap[c] = cat16(*(const v8bf*)&pP[lm * 72 + c * 32 + ka],
                    *(const v8bf*)&pP[lm * 72 + c * 32 + ka + 16]);

    // O += P V : contraction over 64 keys, 4 d-tiles
#pragma unroll
    for (int dt = 0; dt < 4; ++dt) {
      const __bf16* vp = Vp + (size_t)(dt * 16 + lm) * 2048 + t0 + kb2;
#pragma unroll
      for (int c = 0; c < 2; ++c) {
        v16bf bv = cat16(*(const v8bf*)(vp + c * 32),
                         *(const v8bf*)(vp + c * 32 + 16));
        o[dt] = wmma_bf16(ap[c], bv, o[dt]);
      }
    }
  }

  // normalize + store to [B,S,E] bf16
  const int b = bh >> 4, h = bh & 15;
#pragma unroll
  for (int r = 0; r < 8; ++r) {
    float inv = 1.0f / lrow[r];
    int srow = q0 + hi * 8 + r;
    size_t base = ((size_t)b * 2048 + srow) * 1024 + h * 64;
#pragma unroll
    for (int dt = 0; dt < 4; ++dt)
      Ao[base + dt * 16 + lm] = f2bf(o[dt][r] * inv);
  }
}

// ---------------------------------------------------------------------------
// Host launcher
// ---------------------------------------------------------------------------
extern "C" void kernel_launch(void* const* d_in, const int* in_sizes, int n_in,
                              void* d_out, int out_size, void* d_ws,
                              size_t ws_size, hipStream_t stream) {
  const float* x     = (const float*)d_in[0];  // [4,2048,1024]
  const float* w_in  = (const float*)d_in[1];  // [3072,1024]
  const float* b_in  = (const float*)d_in[2];  // [3072]
  const float* w_out = (const float*)d_in[3];  // [1024,1024]
  const float* b_out = (const float*)d_in[4];  // [1024]
  float* out = (float*)d_out;                  // [4,2048,1024]

  char* ws = (char*)d_ws;
  __bf16* xb   = (__bf16*)(ws);                        // 16 MiB  x bf16
  __bf16* wqkv = (__bf16*)(ws + (16ull << 20));        //  6 MiB  Wqkv bf16
  __bf16* wo   = (__bf16*)(ws + (22ull << 20));        //  2 MiB  Wo bf16
  __bf16* qb   = (__bf16*)(ws + (24ull << 20));        // 16 MiB  Q [B,H,S,D]
  __bf16* kb   = (__bf16*)(ws + (40ull << 20));        // 16 MiB  K [B,H,S,D]
  __bf16* vT   = (__bf16*)(ws + (56ull << 20));        // 16 MiB  V [B,H,D,S]
  __bf16* ao   = (__bf16*)(ws + (72ull << 20));        // 16 MiB  attn out [B,S,E]

  cvt_kernel<<<8388608 / 4 / 256, 256, 0, stream>>>(x, xb, 8388608 / 4);
  cvt_kernel<<<3145728 / 4 / 256, 256, 0, stream>>>(w_in, wqkv, 3145728 / 4);
  cvt_kernel<<<1048576 / 4 / 256, 256, 0, stream>>>(w_out, wo, 1048576 / 4);

  gemm_qkv_kernel<<<dim3(32, 24), 256, 0, stream>>>(xb, wqkv, b_in, qb, kb, vT);
  attn_kernel<<<dim3(32, 64), 128, 0, stream>>>(qb, kb, vT, ao);
  gemm_out_kernel<<<dim3(32, 8), 256, 0, stream>>>(ao, wo, b_out, out);
}
